// GNNModel_33088428048866
// MI455X (gfx1250) — matrix operand order repaired and checked by reading, not compiled, not run on hardware
//
#include <hip/hip_runtime.h>

// ---------------------------------------------------------------------------
// GraphSAGE (2 layers, mean aggr) + BN/ReLU + FC for gfx1250 (MI455X).
// GEMMs: v_wmma_f32_16x16x32_bf16, f32 accumulate, 32-row x 32-col tiles per
// wave pair of WMMAs; graph scatter-mean in f32 atomics (L2-resident:
// x=51MB, h=102MB < 192MB L2 -> workload is L2-bandwidth bound).
// ---------------------------------------------------------------------------

typedef __bf16 bf16_t;
typedef __attribute__((ext_vector_type(16))) __bf16 v16bf;
typedef __attribute__((ext_vector_type(8)))  __bf16 v8bf;
typedef __attribute__((ext_vector_type(8)))  float  v8f;

union ABFrag { v16bf v; v8bf h[2]; };

__device__ __forceinline__ v8f wmma_bf16(const ABFrag& a, const ABFrag& b, v8f c) {
    return __builtin_amdgcn_wmma_f32_16x16x32_bf16(false, a.v, false, b.v,
                                                   (short)0, c, false, false);
}

// ---------------------------------------------------------------------------
// Elementwise helpers
// ---------------------------------------------------------------------------
__global__ void zero_f32(float* __restrict__ p, long n) {
    long i = (long)blockIdx.x * blockDim.x + threadIdx.x;
    if (i < n) p[i] = 0.0f;
}

__global__ void cvt_f32_bf16(const float* __restrict__ in, bf16_t* __restrict__ out, long n) {
    long i = (long)blockIdx.x * blockDim.x + threadIdx.x;
    if (i < n) out[i] = (bf16_t)in[i];
}

// deg[dst] += 1 per edge
__global__ void deg_count(const int* __restrict__ dst, float* __restrict__ deg, int E) {
    int i = blockIdx.x * blockDim.x + threadIdx.x;
    if (i < E) atomicAdd(&deg[dst[i]], 1.0f);
}

// One wave32 per edge: gather f32 row of src, atomic-add into agg[dst].
__global__ void scatter_add_f32(const float* __restrict__ feat,
                                const int* __restrict__ src, const int* __restrict__ dst,
                                float* __restrict__ agg, int E, int D) {
    long t = (long)blockIdx.x * blockDim.x + threadIdx.x;
    int e = (int)(t >> 5);
    int lane = (int)(t & 31);
    if (e >= E) return;
    long s = (long)src[e] * D;
    long d = (long)dst[e] * D;
    for (int j = lane; j < D; j += 32)
        atomicAdd(&agg[d + j], feat[s + j]);
}

// Same, but source features are bf16 (halves gather read traffic); accumulate f32.
__global__ void scatter_add_bf16(const bf16_t* __restrict__ feat,
                                 const int* __restrict__ src, const int* __restrict__ dst,
                                 float* __restrict__ agg, int E, int D) {
    long t = (long)blockIdx.x * blockDim.x + threadIdx.x;
    int e = (int)(t >> 5);
    int lane = (int)(t & 31);
    if (e >= E) return;
    long s = (long)src[e] * D;
    long d = (long)dst[e] * D;
    for (int j = lane; j < D; j += 32)
        atomicAdd(&agg[d + j], (float)feat[s + j]);
}

// agg / max(deg,1) -> bf16, shift = log2(D)
__global__ void mean_cvt_bf16(const float* __restrict__ agg, const float* __restrict__ deg,
                              bf16_t* __restrict__ out, long n, int shift) {
    long i = (long)blockIdx.x * blockDim.x + threadIdx.x;
    if (i >= n) return;
    float dg = deg[i >> shift];
    float r = dg > 1.0f ? (1.0f / dg) : 1.0f;
    out[i] = (bf16_t)(agg[i] * r);
}

// Per-channel sum / sumsq over HID=256 columns. Block = 256 threads, one col each.
__global__ void col_stats(const float* __restrict__ h, float* __restrict__ stats,
                          int N, int rows_per_block) {
    int c = threadIdx.x;                 // 0..255
    int r0 = blockIdx.x * rows_per_block;
    float s = 0.0f, s2 = 0.0f;
    for (int rr = 0; rr < rows_per_block; ++rr) {
        int row = r0 + rr;
        if (row < N) {
            float v = h[(long)row * 256 + c];
            s += v; s2 += v * v;
        }
    }
    atomicAdd(&stats[c], s);
    atomicAdd(&stats[256 + c], s2);
}

// BN (population var, eps=1e-5) + ReLU + convert to bf16. HID fixed at 256.
__global__ void bn_relu_cvt(const float* __restrict__ h, const float* __restrict__ stats,
                            const float* __restrict__ g, const float* __restrict__ be,
                            bf16_t* __restrict__ out, long n, float invN) {
    long i = (long)blockIdx.x * blockDim.x + threadIdx.x;
    if (i >= n) return;
    int c = (int)(i & 255);
    float mu  = stats[c] * invN;
    float var = stats[256 + c] * invN - mu * mu;
    float y = (h[i] - mu) * rsqrtf(var + 1e-5f) * g[c] + be[c];
    out[i] = (bf16_t)(y > 0.0f ? y : 0.0f);
}

// ---------------------------------------------------------------------------
// GEMM: C[32 x ncol] = A1*B1 (+ A2*B2 if DUAL) + bias
// A: [N x K] bf16 row-major, B: [K x ncol] bf16 row-major, C: f32.
// Block = (ncol/32)*32 threads; each wave32 owns a 32-column strip and TWO
// 16-row tiles (B fragments reused across both -> 8 straight-line WMMAs per
// k-step for DUAL). DUAL is a template constant so no EXEC manipulation ever
// surrounds the WMMAs. Requires N % 32 == 0 (100000 = 32*3125), K % 32 == 0,
// ncol % 32 == 0 — all exact here, so EXEC is all-1s throughout.
// ---------------------------------------------------------------------------
template <bool DUAL>
__global__ void __launch_bounds__(256)
gemm_wmma(const bf16_t* __restrict__ A1, const bf16_t* __restrict__ A2,
          const bf16_t* __restrict__ B1, const bf16_t* __restrict__ B2,
          const float* __restrict__ bias, float* __restrict__ C,
          int K, int ncol) {
    const int lane = threadIdx.x & 31;
    const int n0   = (threadIdx.x >> 5) << 5;   // wave's 32-column strip
    const int half = lane >> 4;                 // 0: lanes 0-15, 1: lanes 16-31
    const int r    = lane & 15;
    const long row0 = (long)blockIdx.x * 32;

    // A fragment: lane holds row (lane&15); K range {0-7,16-23} (+8 upper half)
    const bf16_t* pa1 = A1 + (row0 + r) * (long)K + half * 8;
    const bf16_t* pa2 = DUAL ? (A2 + (row0 + r) * (long)K + half * 8) : nullptr;
    const long arow16 = 16L * K;                // second 16-row tile

    // B fragment: lane holds column n0+(lane&15); K range {0-7,16-23} (+8 upper half)
    const int  kr    = half << 3;
    const long bstep = 32L * ncol;
    const bf16_t* pb1 = B1 + (long)kr * ncol + n0 + r;
    const bf16_t* pb2 = DUAL ? (B2 + (long)kr * ncol + n0 + r) : nullptr;

    v8f acc00 = {0.f,0.f,0.f,0.f,0.f,0.f,0.f,0.f};   // rows 0-15,  cols n0..n0+15
    v8f acc01 = {0.f,0.f,0.f,0.f,0.f,0.f,0.f,0.f};   // rows 0-15,  cols n0+16..
    v8f acc10 = {0.f,0.f,0.f,0.f,0.f,0.f,0.f,0.f};   // rows 16-31, cols n0..
    v8f acc11 = {0.f,0.f,0.f,0.f,0.f,0.f,0.f,0.f};   // rows 16-31, cols n0+16..

    for (int k0 = 0; k0 < K; k0 += 32) {
        ABFrag bt0, bt1, a0, a1;
        for (int s = 0; s < 16; ++s) {
            long o = (long)(s < 8 ? s : s + 8) * ncol;
            bt0.v[s] = pb1[o];
            bt1.v[s] = pb1[o + 16];
        }
        a0.h[0]  = *(const v8bf*)(pa1);
        a0.h[1]  = *(const v8bf*)(pa1 + 16);
        a1.h[0]  = *(const v8bf*)(pa1 + arow16);
        a1.h[1]  = *(const v8bf*)(pa1 + arow16 + 16);
        acc00 = wmma_bf16(a0, bt0, acc00);
        acc01 = wmma_bf16(a0, bt1, acc01);
        acc10 = wmma_bf16(a1, bt0, acc10);
        acc11 = wmma_bf16(a1, bt1, acc11);
        if (DUAL) {
            ABFrag ct0, ct1, c0, c1;
            for (int s = 0; s < 16; ++s) {
                long o = (long)(s < 8 ? s : s + 8) * ncol;
                ct0.v[s] = pb2[o];
                ct1.v[s] = pb2[o + 16];
            }
            c0.h[0]  = *(const v8bf*)(pa2);
            c0.h[1]  = *(const v8bf*)(pa2 + 16);
            c1.h[0]  = *(const v8bf*)(pa2 + arow16);
            c1.h[1]  = *(const v8bf*)(pa2 + arow16 + 16);
            acc00 = wmma_bf16(c0, ct0, acc00);
            acc01 = wmma_bf16(c0, ct1, acc01);
            acc10 = wmma_bf16(c1, ct0, acc10);
            acc11 = wmma_bf16(c1, ct1, acc11);
            pa2 += 32;
            pb2 += bstep;
        }
        pa1 += 32;
        pb1 += bstep;
    }

    // C layout: lanes 0-15 -> N=lane, M=vgpr; lanes 16-31 -> N=lane-16, M=8+vgpr
    const int c = lane & 15;
    const float bv0 = bias[n0 + c];
    const float bv1 = bias[n0 + 16 + c];
#pragma unroll
    for (int v = 0; v < 8; ++v) {
        long ra = row0 + half * 8 + v;        // first 16-row tile
        long rb = ra + 16;                    // second 16-row tile
        C[ra * (long)ncol + n0 + c]      = acc00[v] + bv0;
        C[ra * (long)ncol + n0 + 16 + c] = acc01[v] + bv1;
        C[rb * (long)ncol + n0 + c]      = acc10[v] + bv0;
        C[rb * (long)ncol + n0 + 16 + c] = acc11[v] + bv1;
    }
}

// ---------------------------------------------------------------------------
// Host launcher
// ---------------------------------------------------------------------------
static inline unsigned cdiv(long n, long d) { return (unsigned)((n + d - 1) / d); }

extern "C" void kernel_launch(void* const* d_in, const int* in_sizes, int n_in,
                              void* d_out, int out_size, void* d_ws, size_t ws_size,
                              hipStream_t stream) {
    const float* x    = (const float*)d_in[0];
    const int*   esrc = (const int*)  d_in[1];
    const int*   edst = (const int*)  d_in[2];
    const float* W1l  = (const float*)d_in[3];
    const float* b1   = (const float*)d_in[4];
    const float* W1r  = (const float*)d_in[5];
    const float* g1   = (const float*)d_in[6];
    const float* be1  = (const float*)d_in[7];
    const float* W2l  = (const float*)d_in[8];
    const float* b2   = (const float*)d_in[9];
    const float* W2r  = (const float*)d_in[10];
    const float* g2   = (const float*)d_in[11];
    const float* be2  = (const float*)d_in[12];
    const float* Wfc  = (const float*)d_in[13];
    const float* bfc  = (const float*)d_in[14];

    const int IN = 128, HID = 256, NCLS = 32;
    const long N = (long)in_sizes[0] / IN;      // 100000
    const int  E = in_sizes[1];                 // 1600000

    // Carve workspace (256B-aligned regions)
    size_t off = 0;
    char* base = (char*)d_ws;
    auto carve = [&](size_t bytes) -> void* {
        void* p = base + off;
        off += (bytes + 255) & ~(size_t)255;
        return p;
    };
    float*  deg    = (float*) carve((size_t)N * 4);
    float*  agg    = (float*) carve((size_t)N * HID * 4);   // f32 scatter accumulator
    bf16_t* agg_bf = (bf16_t*)carve((size_t)N * HID * 2);
    bf16_t* x_bf   = (bf16_t*)carve((size_t)N * IN * 2);
    float*  h_pre  = (float*) carve((size_t)N * HID * 4);   // pre-BN activations
    bf16_t* h_bf   = (bf16_t*)carve((size_t)N * HID * 2);   // post-BN/ReLU bf16
    float*  stats  = (float*) carve(512 * 4);               // [sum(256) | sumsq(256)]
    bf16_t* w1l_bf = (bf16_t*)carve((size_t)IN * HID * 2);
    bf16_t* w1r_bf = (bf16_t*)carve((size_t)IN * HID * 2);
    bf16_t* w2l_bf = (bf16_t*)carve((size_t)HID * HID * 2);
    bf16_t* w2r_bf = (bf16_t*)carve((size_t)HID * HID * 2);
    bf16_t* wfc_bf = (bf16_t*)carve((size_t)HID * NCLS * 2);
    (void)ws_size; (void)n_in; (void)out_size;

    const int T = 256;
    const unsigned rowBlks = (unsigned)(N / 32);            // 3125 (32 rows/block)
    const unsigned scatterBlks = cdiv((long)E * 32, T);     // one wave per edge

    // --- One-time conversions ---
    cvt_f32_bf16<<<cdiv(N * IN, T), T, 0, stream>>>(x, x_bf, N * IN);
    cvt_f32_bf16<<<cdiv((long)IN * HID, T), T, 0, stream>>>(W1l, w1l_bf, (long)IN * HID);
    cvt_f32_bf16<<<cdiv((long)IN * HID, T), T, 0, stream>>>(W1r, w1r_bf, (long)IN * HID);
    cvt_f32_bf16<<<cdiv((long)HID * HID, T), T, 0, stream>>>(W2l, w2l_bf, (long)HID * HID);
    cvt_f32_bf16<<<cdiv((long)HID * HID, T), T, 0, stream>>>(W2r, w2r_bf, (long)HID * HID);
    cvt_f32_bf16<<<cdiv((long)HID * NCLS, T), T, 0, stream>>>(Wfc, wfc_bf, (long)HID * NCLS);

    // --- Degrees (shared by both layers) ---
    zero_f32<<<cdiv(N, T), T, 0, stream>>>(deg, N);
    deg_count<<<cdiv(E, T), T, 0, stream>>>(edst, deg, E);

    // ================= Layer 1 =================
    zero_f32<<<cdiv(N * IN, T), T, 0, stream>>>(agg, N * IN);
    scatter_add_f32<<<scatterBlks, T, 0, stream>>>(x, esrc, edst, agg, E, IN);
    mean_cvt_bf16<<<cdiv(N * IN, T), T, 0, stream>>>(agg, deg, agg_bf, N * IN, 7);

    gemm_wmma<true><<<rowBlks, 256, 0, stream>>>(agg_bf, x_bf, w1l_bf, w1r_bf,
                                                 b1, h_pre, IN, HID);

    zero_f32<<<cdiv(512, T), T, 0, stream>>>(stats, 512);
    col_stats<<<cdiv(N, 128), 256, 0, stream>>>(h_pre, stats, (int)N, 128);
    bn_relu_cvt<<<cdiv(N * HID, T), T, 0, stream>>>(h_pre, stats, g1, be1, h_bf,
                                                    N * HID, 1.0f / (float)N);

    // ================= Layer 2 =================
    zero_f32<<<cdiv(N * HID, T), T, 0, stream>>>(agg, N * HID);
    scatter_add_bf16<<<scatterBlks, T, 0, stream>>>(h_bf, esrc, edst, agg, E, HID);
    mean_cvt_bf16<<<cdiv(N * HID, T), T, 0, stream>>>(agg, deg, agg_bf, N * HID, 8);

    gemm_wmma<true><<<rowBlks, 256, 0, stream>>>(agg_bf, h_bf, w2l_bf, w2r_bf,
                                                 b2, h_pre, HID, HID);

    zero_f32<<<cdiv(512, T), T, 0, stream>>>(stats, 512);
    col_stats<<<cdiv(N, 128), 256, 0, stream>>>(h_pre, stats, (int)N, 128);
    bn_relu_cvt<<<cdiv(N * HID, T), T, 0, stream>>>(h_pre, stats, g2, be2, h_bf,
                                                    N * HID, 1.0f / (float)N);

    // ================= FC head =================
    // out[N x 32] = h @ Wfc + bfc : one wave per 32-row block (ncol = 32)
    gemm_wmma<false><<<rowBlks, 32, 0, stream>>>(h_bf, nullptr, wfc_bf, nullptr,
                                                 bfc, (float*)d_out, HID, NCLS);
}
